// MD17HybridHCNet_13950053777560
// MI455X (gfx1250) — matrix operand order repaired and checked by reading, not compiled, not run on hardware
//
#include <hip/hip_runtime.h>

#define Bb 1024
#define Nn 64
#define Hh 128
#define Ll 4
#define Kk 16
#define Rr 20
#define CUTOFF 5.0f
#define GAMMA 10.0f
#define APB 4      // atoms per message block
#define MK 288     // 276 padded to 9*32

typedef __attribute__((ext_vector_type(16))) __bf16 v16bf;
typedef __attribute__((ext_vector_type(8)))  float  v8f;

union Frag16 { v16bf bf; unsigned int w[8]; };
union CFrag  { v8f v; float f[8]; };

static __device__ __forceinline__ unsigned short f32_to_bf16(float x) {
    union { float f; unsigned u; } v; v.f = x;
    unsigned r = v.u + 0x7FFFu + ((v.u >> 16) & 1u);
    return (unsigned short)(r >> 16);
}
// fast silu: x * v_rcp_f32(1+exp(-x)) -- avoids IEEE div_scale fixup chain
static __device__ __forceinline__ float dev_silu(float x) {
    return x * __builtin_amdgcn_rcpf(1.0f + __expf(-x));
}

// ---- WMMA fragment loaders -------------------------------------------------
// A tile (16x32 bf16) from row-major LDS [rows][ldk]; 16-bit A layout:
// lanes 0-15 rows M=0..15 hold K pairs {0..7,16..23}; lanes 16-31 hold {8..15,24..31}
static __device__ __forceinline__ Frag16 load_a(const unsigned short* lds,
                                                int row0, int k0, int ldk, int lane) {
    Frag16 f;
    int row = row0 + (lane & 15);
    int g   = (lane >> 4) << 3;               // 0 or 8
    const unsigned short* p = lds + (size_t)row * ldk + k0;
#pragma unroll
    for (int j = 0; j < 8; j++) {
        int kb = ((j & 4) ? 16 : 0) + g + ((j & 3) << 1);
        f.w[j] = *(const unsigned int*)(p + kb);
    }
    return f;
}
// B tile pre-packed in lane order: tile = 512 bf16, lane's 16 values contiguous
static __device__ __forceinline__ Frag16 load_b(const unsigned short* Wpk,
                                                int tileIdx, int lane) {
    Frag16 f;
    const uint4* p = (const uint4*)(Wpk + ((size_t)tileIdx << 9) + (lane << 4));
    uint4 a = p[0], b = p[1];
    f.w[0]=a.x; f.w[1]=a.y; f.w[2]=a.z; f.w[3]=a.w;
    f.w[4]=b.x; f.w[5]=b.y; f.w[6]=b.z; f.w[7]=b.w;
    return f;
}

// ---- Weight packing: [K][N] f32 -> bf16 WMMA-B fragment order --------------
__global__ void pack_b_kernel(const float* __restrict__ src, unsigned short* __restrict__ dst,
                              int Kdim, int Ndim, int ntiles) {
    int tile = blockIdx.x;
    const float* s = src + (size_t)blockIdx.y * Kdim * Ndim;
    unsigned short* d = dst + ((size_t)blockIdx.y * gridDim.x + tile) * 512;
    int kt = tile / ntiles, nt = tile % ntiles;
    for (int e = threadIdx.x; e < 512; e += blockDim.x) {
        int lane = e >> 4, p = e & 15;
        int k = kt * 32 + ((p & 8) << 1) + ((lane >> 4) << 3) + (p & 7);   // B layout: k = (p/8)*16 + (lane/16)*8 + p%8
        int n = nt * 16 + (lane & 15);
        float v = (k < Kdim && n < Ndim) ? s[(size_t)k * Ndim + n] : 0.0f;
        d[e] = f32_to_bf16(v);
    }
}

// ---- Embedding + kNN + RBF/envelope ----------------------------------------
__global__ void embed_knn_kernel(const float* __restrict__ positions,
                                 const int* __restrict__ zatoms,
                                 const float* __restrict__ atom_embed,
                                 const float* __restrict__ pos_W,
                                 const float* __restrict__ pos_b,
                                 float* __restrict__ h, unsigned short* __restrict__ hbf,
                                 int* __restrict__ idx, unsigned short* __restrict__ rbfb,
                                 float* __restrict__ env) {
    __shared__ float pos[Nn][3];
    int b = blockIdx.x, n = threadIdx.x;
    for (int t = n; t < Nn * 3; t += blockDim.x)
        ((float*)pos)[t] = positions[(size_t)b * Nn * 3 + t];
    __syncthreads();
    int z = zatoms[(size_t)b * Nn + n];
    float px = pos[n][0], py = pos[n][1], pz = pos[n][2];
    size_t hbase = ((size_t)b * Nn + n) * Hh;
    for (int c = 0; c < Hh; c++) {
        float v;
        if (c < 32) v = atom_embed[z * 32 + c];
        else {
            int cc = c - 32;
            v = pos_b[cc] + px * pos_W[cc] + py * pos_W[96 + cc] + pz * pos_W[192 + cc];
        }
        h[hbase + c] = v;
        hbf[hbase + c] = f32_to_bf16(v);
    }
    float d2[Nn];
    for (int j = 0; j < Nn; j++) {
        float dx = px - pos[j][0], dy = py - pos[j][1], dz = pz - pos[j][2];
        d2[j] = dx * dx + dy * dy + dz * dz;
    }
    d2[n] = 1e9f;
    size_t ibase = ((size_t)b * Nn + n) * Kk;
    for (int k = 0; k < Kk; k++) {
        int best = 0; float bd = 1e30f;
        for (int j = 0; j < Nn; j++) if (d2[j] < bd) { bd = d2[j]; best = j; }
        d2[best] = 1e30f;
        idx[ibase + k] = best;
        float dist = sqrtf(fmaxf(bd, 1e-12f));
        for (int r = 0; r < Rr; r++) {
            float ctr = CUTOFF * (float)r / (float)(Rr - 1);
            float t = dist - ctr;
            rbfb[(ibase + k) * Rr + r] = f32_to_bf16(__expf(-GAMMA * t * t));
        }
        env[ibase + k] = (dist < CUTOFF) ? 0.5f * (__cosf(3.14159265358979f * dist / CUTOFF) + 1.0f) : 0.0f;
    }
}

// ---- Message MLP: per atom [16 edges x 288] @ [288 x 128], silu*env, row-sum
__global__ void message_kernel(const unsigned short* __restrict__ hbf,
                               const int* __restrict__ idx,
                               const unsigned short* __restrict__ rbfb,
                               const float* __restrict__ env,
                               const unsigned short* __restrict__ msgPk,
                               const float* __restrict__ msg_bL,
                               float* __restrict__ msum) {
    __shared__ unsigned short A[APB * 16][MK];     // 64 x 288 bf16 = 36 KB
    __shared__ float envs[APB * 16];
    int b = blockIdx.x >> 4;
    int a0 = (blockIdx.x & 15) << 2;               // 4 atoms per block
    int tid = threadIdx.x;
    {
        int row = tid >> 2, q = tid & 3;           // 4 threads per edge-row
        int i = a0 + (row >> 4), e = row & 15;
        size_t ebase = ((size_t)(b * Nn) + i) * Kk + e;
        int j = idx[ebase];
        const unsigned short* hi = hbf + ((size_t)(b * Nn) + i) * Hh;
        const unsigned short* hj = hbf + ((size_t)(b * Nn) + j) * Hh;
        // cols 0..255 : vectorized b128 copy, 64 cols (=8 uint4) per thread
        const uint4* s4 = (const uint4*)((q < 2) ? (hi + q * 64) : (hj + (q - 2) * 64));
        uint4* d4 = (uint4*)(&A[row][q * 64]);
#pragma unroll
        for (int x = 0; x < 8; x++) d4[x] = s4[x];
        // cols 256..287 : rbf (20) + zero pad, 8 cols per thread
        for (int c = 256 + q * 8; c < 256 + q * 8 + 8; c++)
            A[row][c] = (c < 276) ? rbfb[ebase * Rr + (c - 256)] : (unsigned short)0;
        if (tid < APB * 16)
            envs[tid] = env[((size_t)(b * Nn) + a0 + (tid >> 4)) * Kk + (tid & 15)];
    }
    __syncthreads();
    int wid = tid >> 5, lane = tid & 31;
    for (int t = wid; t < 4 * 8; t += 8) {
        int mt = t >> 3, nt = t & 7;               // mt == local atom (16 edges)
        CFrag c;
#pragma unroll
        for (int r = 0; r < 8; r++) c.f[r] = 0.0f;
        for (int kt = 0; kt < 9; kt++) {
            Frag16 af = load_a(&A[0][0], mt * 16, kt * 32, MK, lane);
            Frag16 bf = load_b(msgPk, kt * 8 + nt, lane);
            c.v = __builtin_amdgcn_wmma_f32_16x16x32_bf16(false, af.bf, false, bf.bf,
                                                          (short)0, c.v, false, false);
        }
        float bias = msg_bL[nt * 16 + (lane & 15)];
        float part = 0.0f;
#pragma unroll
        for (int r = 0; r < 8; r++) {
            int M = ((lane >> 4) << 3) + r;        // edge index within atom
            float m = dev_silu(c.f[r] + bias);
            part += m * envs[mt * 16 + M];
        }
        float tot = part + __shfl_xor(part, 16);
        if (lane < 16)
            msum[((size_t)(b * Nn) + a0 + mt) * Hh + nt * 16 + lane] = tot;
    }
}

// ---- Generic row GEMM: out = [res +] act( [X0|X1] @ Wpk + bias [+ addp] ) --
__global__ void gemm_kernel(const float* __restrict__ X0, const float* __restrict__ X1,
                            int k0, int k1,
                            const unsigned short* __restrict__ Wpk,
                            const float* __restrict__ bias,
                            const float* __restrict__ addp, int addDiv,
                            const float* __restrict__ res,
                            float* __restrict__ out,
                            int rows, int Hout, int outCols, int act) {
    extern __shared__ unsigned short A[];          // [32][Kdim] bf16
    int Kdim = k0 + k1;
    int row0 = blockIdx.x * 32;
    int tid = threadIdx.x;
    // vectorized fill: float4 load -> packed bf16x4 (uint2) store
    int kq = Kdim >> 2;
    for (int e = tid; e < 32 * kq; e += blockDim.x) {
        int r = e / kq, cq = (e % kq) << 2;
        int row = row0 + r;
        float4 v = make_float4(0.f, 0.f, 0.f, 0.f);
        if (row < rows) {
            const float* src = (cq < k0) ? (X0 + (size_t)row * k0 + cq)
                                         : (X1 + (size_t)row * k1 + (cq - k0));
            v = *(const float4*)src;
        }
        unsigned p0 = (unsigned)f32_to_bf16(v.x) | ((unsigned)f32_to_bf16(v.y) << 16);
        unsigned p1 = (unsigned)f32_to_bf16(v.z) | ((unsigned)f32_to_bf16(v.w) << 16);
        *(uint2*)(A + r * Kdim + cq) = make_uint2(p0, p1);
    }
    __syncthreads();
    int wid = tid >> 5, lane = tid & 31;
    int ntiles = Hout >> 4, ksteps = Kdim >> 5;
    for (int t = wid; t < 2 * ntiles; t += (blockDim.x >> 5)) {
        int mt = t / ntiles, nt = t % ntiles;
        CFrag c;
#pragma unroll
        for (int r = 0; r < 8; r++) c.f[r] = 0.0f;
        for (int kt = 0; kt < ksteps; kt++) {
            Frag16 af = load_a(A, mt * 16, kt * 32, Kdim, lane);
            Frag16 bf = load_b(Wpk, kt * ntiles + nt, lane);
            c.v = __builtin_amdgcn_wmma_f32_16x16x32_bf16(false, af.bf, false, bf.bf,
                                                          (short)0, c.v, false, false);
        }
        int ncol = nt * 16 + (lane & 15);
        if (ncol >= outCols) continue;
        float bv = bias ? bias[ncol] : 0.0f;
#pragma unroll
        for (int r = 0; r < 8; r++) {
            int row = row0 + mt * 16 + ((lane >> 4) << 3) + r;
            if (row >= rows) continue;
            float v = c.f[r] + bv;
            if (addp) v += addp[(size_t)(row / addDiv) * Hout + ncol];
            if (act)  v = dev_silu(v);
            if (res)  v += res[(size_t)row * outCols + ncol];
            out[(size_t)row * outCols + ncol] = v;
        }
    }
}

// ---- Global mean over atoms -------------------------------------------------
__global__ void mean_kernel(const float* __restrict__ h, float* __restrict__ hmean) {
    int b = blockIdx.x, c = threadIdx.x;
    float s = 0.0f;
    for (int n = 0; n < Nn; n++) s += h[((size_t)b * Nn + n) * Hh + c];
    hmean[(size_t)b * Hh + c] = s * (1.0f / (float)Nn);
}

// ---- Clifford block: LN -> W1(512) silu (LDS) -> W2 -> residual, h + hbf ---
__global__ void clifford_kernel(const float* __restrict__ fused,
                                const unsigned short* __restrict__ cb1Pk,
                                const float* __restrict__ cb_b1L,
                                const unsigned short* __restrict__ cb2Pk,
                                const float* __restrict__ cb_b2L,
                                const float* __restrict__ lng,
                                const float* __restrict__ lnb,
                                float* __restrict__ h, unsigned short* __restrict__ hbf) {
    __shared__ unsigned short X[32][Hh];           // LN'd rows, bf16, 8 KB
    __shared__ unsigned short T[32][512];          // silu intermediate, 32 KB
    int row0 = blockIdx.x * 32;
    int tid = threadIdx.x;
    // LayerNorm: 8 lanes per row, single pass (sum + sumsq), shfl tree reduce
    {
        int r = tid >> 3, sub = tid & 7;
        const float* fr = fused + ((size_t)row0 + r) * Hh;
        float s = 0.0f, ss = 0.0f;
        for (int c = sub * 16; c < sub * 16 + 16; c++) {
            float x = fr[c]; s += x; ss += x * x;
        }
#pragma unroll
        for (int d = 1; d < 8; d <<= 1) { s += __shfl_xor(s, d); ss += __shfl_xor(ss, d); }
        float m  = s * (1.0f / Hh);
        float vv = ss * (1.0f / Hh) - m * m;
        float is = rsqrtf(vv + 1e-5f);
        for (int c = sub * 16; c < sub * 16 + 16; c++)
            X[r][c] = f32_to_bf16((fr[c] - m) * is * lng[c] + lnb[c]);
    }
    __syncthreads();
    int wid = tid >> 5, lane = tid & 31;
    // GEMM1: [32x128] @ [128x512]
    for (int t = wid; t < 2 * 32; t += 8) {
        int mt = t >> 5, nt = t & 31;
        CFrag c;
#pragma unroll
        for (int r = 0; r < 8; r++) c.f[r] = 0.0f;
        for (int kt = 0; kt < 4; kt++) {
            Frag16 af = load_a(&X[0][0], mt * 16, kt * 32, Hh, lane);
            Frag16 bf = load_b(cb1Pk, kt * 32 + nt, lane);
            c.v = __builtin_amdgcn_wmma_f32_16x16x32_bf16(false, af.bf, false, bf.bf,
                                                          (short)0, c.v, false, false);
        }
        int ncol = nt * 16 + (lane & 15);
        float bv = cb_b1L[ncol];
#pragma unroll
        for (int r = 0; r < 8; r++) {
            int M = mt * 16 + ((lane >> 4) << 3) + r;
            T[M][ncol] = f32_to_bf16(dev_silu(c.f[r] + bv));
        }
    }
    __syncthreads();
    // GEMM2: [32x512] @ [512x128] + fused residual
    for (int t = wid; t < 2 * 8; t += 8) {
        int mt = t >> 3, nt = t & 7;
        CFrag c;
#pragma unroll
        for (int r = 0; r < 8; r++) c.f[r] = 0.0f;
        for (int kt = 0; kt < 16; kt++) {
            Frag16 af = load_a(&T[0][0], mt * 16, kt * 32, 512, lane);
            Frag16 bf = load_b(cb2Pk, kt * 8 + nt, lane);
            c.v = __builtin_amdgcn_wmma_f32_16x16x32_bf16(false, af.bf, false, bf.bf,
                                                          (short)0, c.v, false, false);
        }
        int ncol = nt * 16 + (lane & 15);
        float bv = cb_b2L[ncol];
#pragma unroll
        for (int r = 0; r < 8; r++) {
            size_t row = (size_t)row0 + mt * 16 + ((lane >> 4) << 3) + r;
            float v = fused[row * Hh + ncol] + c.f[r] + bv;
            h[row * Hh + ncol] = v;
            hbf[row * Hh + ncol] = f32_to_bf16(v);
        }
    }
}

// ---------------------------------------------------------------------------
extern "C" void kernel_launch(void* const* d_in, const int* in_sizes, int n_in,
                              void* d_out, int out_size, void* d_ws, size_t ws_size,
                              hipStream_t stream) {
    (void)in_sizes; (void)n_in; (void)out_size; (void)ws_size;
    const float* positions  = (const float*)d_in[0];
    const int*   zatoms     = (const int*)d_in[1];
    const float* atom_embed = (const float*)d_in[2];
    const float* pos_W      = (const float*)d_in[3];
    const float* pos_bp     = (const float*)d_in[4];
    const float* msg_W      = (const float*)d_in[5];
    const float* msg_b      = (const float*)d_in[6];
    const float* upd_W      = (const float*)d_in[7];
    const float* upd_b      = (const float*)d_in[8];
    const float* gself_W    = (const float*)d_in[9];
    const float* gmean_W    = (const float*)d_in[10];
    const float* g_b        = (const float*)d_in[11];
    const float* fus_W      = (const float*)d_in[12];
    const float* fus_b      = (const float*)d_in[13];
    const float* ln_g       = (const float*)d_in[14];
    const float* ln_b       = (const float*)d_in[15];
    const float* cb_W1      = (const float*)d_in[16];
    const float* cb_b1      = (const float*)d_in[17];
    const float* cb_W2      = (const float*)d_in[18];
    const float* cb_b2      = (const float*)d_in[19];
    const float* fo_W1      = (const float*)d_in[20];
    const float* fo_b1      = (const float*)d_in[21];
    const float* fo_W2      = (const float*)d_in[22];
    const float* fo_b2      = (const float*)d_in[23];
    float* out = (float*)d_out;

    const size_t BN = (size_t)Bb * Nn;
    char* wsb = (char*)d_ws;
    size_t off = 0;
#define WS_ALLOC(T, name, count) \
    T* name = (T*)(wsb + off); off = (off + (size_t)(count) * sizeof(T) + 255) & ~(size_t)255;

    WS_ALLOC(float,          hbuf,  BN * Hh)
    WS_ALLOC(unsigned short, hbf,   BN * Hh)
    WS_ALLOC(int,            idxb,  BN * Kk)
    WS_ALLOC(unsigned short, rbfb,  BN * Kk * Rr)
    WS_ALLOC(float,          envb,  BN * Kk)
    WS_ALLOC(float,          bufA,  BN * Hh)   // msum -> fused -> t1
    WS_ALLOC(float,          bufB,  BN * Hh)   // local
    WS_ALLOC(float,          bufC,  BN * Hh)   // glob
    WS_ALLOC(float,          hmean, (size_t)Bb * Hh)
    WS_ALLOC(float,          gm,    (size_t)Bb * Hh)
    WS_ALLOC(unsigned short, msgPk,   (size_t)Ll * 9 * 8 * 512)
    WS_ALLOC(unsigned short, updPk,   (size_t)Ll * 4 * 8 * 512)
    WS_ALLOC(unsigned short, gselfPk, (size_t)Ll * 4 * 8 * 512)
    WS_ALLOC(unsigned short, gmeanPk, (size_t)Ll * 4 * 8 * 512)
    WS_ALLOC(unsigned short, fusPk,   (size_t)Ll * 8 * 8 * 512)
    WS_ALLOC(unsigned short, cb1Pk,   (size_t)Ll * 4 * 32 * 512)
    WS_ALLOC(unsigned short, cb2Pk,   (size_t)Ll * 16 * 8 * 512)
    WS_ALLOC(unsigned short, fo1Pk,   (size_t)4 * 8 * 512)
    WS_ALLOC(unsigned short, fo2Pk,   (size_t)4 * 1 * 512)
#undef WS_ALLOC

    // --- pack weights to WMMA B-fragment order (bf16) ---
    pack_b_kernel<<<dim3(9 * 8,  Ll), 256, 0, stream>>>(msg_W,   msgPk,   276, 128, 8);
    pack_b_kernel<<<dim3(4 * 8,  Ll), 256, 0, stream>>>(upd_W,   updPk,   128, 128, 8);
    pack_b_kernel<<<dim3(4 * 8,  Ll), 256, 0, stream>>>(gself_W, gselfPk, 128, 128, 8);
    pack_b_kernel<<<dim3(4 * 8,  Ll), 256, 0, stream>>>(gmean_W, gmeanPk, 128, 128, 8);
    pack_b_kernel<<<dim3(8 * 8,  Ll), 256, 0, stream>>>(fus_W,   fusPk,   256, 128, 8);
    pack_b_kernel<<<dim3(4 * 32, Ll), 256, 0, stream>>>(cb_W1,   cb1Pk,   128, 512, 32);
    pack_b_kernel<<<dim3(16 * 8, Ll), 256, 0, stream>>>(cb_W2,   cb2Pk,   512, 128, 8);
    pack_b_kernel<<<dim3(4 * 8,  1),  256, 0, stream>>>(fo_W1,   fo1Pk,   128, 128, 8);
    pack_b_kernel<<<dim3(4 * 1,  1),  256, 0, stream>>>(fo_W2,   fo2Pk,   128, 3,   1);

    // --- embeddings + kNN graph ---
    embed_knn_kernel<<<Bb, Nn, 0, stream>>>(positions, zatoms, atom_embed, pos_W, pos_bp,
                                            hbuf, hbf, idxb, rbfb, envb);

    int gemmBlocks = (int)(BN / 32);
    for (int l = 0; l < Ll; l++) {
        // message MLP -> msum (bufA)
        message_kernel<<<Bb * (Nn / APB), 256, 0, stream>>>(
            hbf, idxb, rbfb, envb,
            msgPk + (size_t)l * 9 * 8 * 512, msg_b + (size_t)l * Hh, bufA);
        // local = h + silu(msum @ upd_W + b)
        gemm_kernel<<<gemmBlocks, 256, 32 * 128 * 2, stream>>>(
            bufA, nullptr, 128, 0, updPk + (size_t)l * 4 * 8 * 512,
            upd_b + (size_t)l * Hh, nullptr, 1, hbuf, bufB, (int)BN, 128, 128, 1);
        // global mean-field
        mean_kernel<<<Bb, Hh, 0, stream>>>(hbuf, hmean);
        gemm_kernel<<<Bb / 32, 256, 32 * 128 * 2, stream>>>(
            hmean, nullptr, 128, 0, gmeanPk + (size_t)l * 4 * 8 * 512,
            nullptr, nullptr, 1, nullptr, gm, Bb, 128, 128, 0);
        gemm_kernel<<<gemmBlocks, 256, 32 * 128 * 2, stream>>>(
            hbuf, nullptr, 128, 0, gselfPk + (size_t)l * 4 * 8 * 512,
            g_b + (size_t)l * Hh, gm, Nn, nullptr, bufC, (int)BN, 128, 128, 1);
        // fusion: silu([local|glob] @ fus_W + b) -> bufA
        gemm_kernel<<<gemmBlocks, 256, 32 * 256 * 2, stream>>>(
            bufB, bufC, 128, 128, fusPk + (size_t)l * 8 * 8 * 512,
            fus_b + (size_t)l * Hh, nullptr, 1, nullptr, bufA, (int)BN, 128, 128, 1);
        // clifford residual MLP -> h, hbf
        clifford_kernel<<<gemmBlocks, 256, 0, stream>>>(
            bufA, cb1Pk + (size_t)l * 4 * 32 * 512, cb_b1 + (size_t)l * 512,
            cb2Pk + (size_t)l * 16 * 8 * 512, cb_b2 + (size_t)l * Hh,
            ln_g + (size_t)l * Hh, ln_b + (size_t)l * Hh, hbuf, hbf);
    }
    // force head
    gemm_kernel<<<gemmBlocks, 256, 32 * 128 * 2, stream>>>(
        hbuf, nullptr, 128, 0, fo1Pk, fo_b1, nullptr, 1, nullptr, bufA, (int)BN, 128, 128, 1);
    gemm_kernel<<<gemmBlocks, 256, 32 * 128 * 2, stream>>>(
        bufA, nullptr, 128, 0, fo2Pk, fo_b2, nullptr, 1, nullptr, out, (int)BN, 16, 3, 0);
}
